// EnerGDev_58360015618571
// MI455X (gfx1250) — compile-verified, matrix operand order checked by reading, not run on hardware
//
#include <hip/hip_runtime.h>
#include <hip/hip_bf16.h>

typedef __bf16 bf16_t;
typedef __attribute__((ext_vector_type(16))) __bf16 v16bf;
typedef __attribute__((ext_vector_type(8)))  float  v8f;

#define AM_EDGE        0
#define AM_F32         1
#define AM_BF16        2
#define AM_BF16_BNACT  3
#define EP_BF16_STATS  0
#define EP_BF16_ACT    1
#define EP_F32_STATS   2

__device__ __forceinline__ float lsilu(float x, float a) {
    // SiLU(x) + a*x
    return x / (1.0f + __expf(-x)) + a * x;
}

// ---------------------------------------------------------------------------
// Pack fp32 weights [Kact x N] row-major into WMMA B-fragment order (bf16):
// linear index = ((kb*N + n)*2 + half)*16 + e,  K = kb*32 + (e&7) + 16*(e>>3) + 8*half
// Rows k >= Kact are zero (K padding for conv1's K=8 case).
// ---------------------------------------------------------------------------
__global__ void k_pack(const float* __restrict__ W, bf16_t* __restrict__ Wp,
                       int Kact, int Kpad, int N) {
    int i = blockIdx.x * blockDim.x + threadIdx.x;
    int total = Kpad * N;                    // (Kpad/32)*N*32
    if (i >= total) return;
    int e    = i & 15;
    int half = (i >> 4) & 1;
    int rest = i >> 5;
    int n    = rest % N;
    int kb   = rest / N;
    int k    = (kb << 5) + (e & 7) + ((e >> 3) << 4) + (half << 3);
    float v  = (k < Kact) ? W[(long)k * N + n] : 0.0f;
    Wp[i] = (bf16_t)v;
}

// ---------------------------------------------------------------------------
// Node prep: nf[:,0] = x[:,0]; nf[:,1:4] = x[:,1:4] @ matrix[batch_index]
// nf row stride is fixed at 128.
// ---------------------------------------------------------------------------
__global__ void k_prep(const float* __restrict__ x, const float* __restrict__ mat,
                       const int* __restrict__ bidx, float* __restrict__ nf, long NN) {
    long n = (long)blockIdx.x * blockDim.x + threadIdx.x;
    if (n >= NN) return;
    const float* Mg = mat + (long)bidx[n] * 9;
    float f0 = x[n * 4 + 1], f1 = x[n * 4 + 2], f2 = x[n * 4 + 3];
    nf[n * 128 + 0] = x[n * 4 + 0];
#pragma unroll
    for (int k = 0; k < 3; ++k)
        nf[n * 128 + 1 + k] = f0 * Mg[k] + f1 * Mg[3 + k] + f2 * Mg[6 + k];
}

__global__ void k_degree(const int* __restrict__ dst, float* __restrict__ deg, long E) {
    long e = (long)blockIdx.x * blockDim.x + threadIdx.x;
    if (e >= E) return;
    atomicAdd(&deg[dst[e]], 1.0f);
}

// ---------------------------------------------------------------------------
// WMMA GEMM:  C[M,N] = A[M,K] @ Wp[K,N] + bias, with pluggable A sources and
// epilogues. Block = 4 waves; covers 128 rows x 64 cols; K multiple of 32.
// A staged to LDS in fragment order; block's weight slab cached in LDS.
// ---------------------------------------------------------------------------
__global__ __launch_bounds__(128) void k_gemm(
    int mode_a, int epi,
    const float* __restrict__ Af, const bf16_t* __restrict__ Ab, int lda, int kact,
    const int* __restrict__ esrc, const int* __restrict__ edst,
    const float* __restrict__ iscale, const float* __restrict__ ishift,
    const bf16_t* __restrict__ Wp, const float* __restrict__ bias,
    bf16_t* __restrict__ Cb, float* __restrict__ Cf, int ldc,
    float* __restrict__ ssum, float* __restrict__ ssq,
    long M, int K, int N)
{
    extern __shared__ v16bf smem_v[];
    bf16_t* a_lds = (bf16_t*)smem_v;           // K*16 bf16 (<=16KB)
    bf16_t* w_lds = a_lds + (long)K * 16;      // K*64 bf16 (<=64KB)

    const int tid  = threadIdx.x;
    const int wave = tid >> 5;
    const int lane = tid & 31;
    const int KB   = K >> 5;
    const long rows0 = (long)blockIdx.x * 128;
    const int n0   = blockIdx.y * 64;
    const int arow = lane & 15;
    const int ahalf = lane >> 4;
    const int nl_lane = (wave << 4) + arow;    // local col in [0,64)
    const int col  = n0 + nl_lane;
    const bool active = (col < N);

    // ---- stage packed-weight slab for this block's 64 columns into LDS ----
    {
        int chunks = K * 8;                    // K*64 elems / 8 per uint4
        for (int c = tid; c < chunks; c += 128) {
            int lin  = c << 3;
            int e0   = lin & 15;
            int half = (lin >> 4) & 1;
            int nl   = (lin >> 5) & 63;
            int kb   = lin >> 11;
            uint4 v  = make_uint4(0u, 0u, 0u, 0u);
            int n = n0 + nl;
            if (n < N)
                v = *(const uint4*)(Wp + ((((long)kb * N + n) * 2 + half) << 4) + e0);
            *(uint4*)(w_lds + lin) = v;
        }
    }
    __syncthreads();

    float lsum = 0.0f, lsq = 0.0f;
    const float bcol = active ? bias[col] : 0.0f;

    for (int ms = 0; ms < 8; ++ms) {
        const long m0 = rows0 + ms * 16;

        // ---- stage A tile (16 rows x K) into LDS in fragment order ----
        const int tot = K * 16;
        for (int i = tid; i < tot; i += 128) {
            int e    = i & 15;
            int half = (i >> 4) & 1;
            int m    = (i >> 5) & 15;
            int kb   = i >> 9;
            int k    = (kb << 5) + (e & 7) + ((e >> 3) << 4) + (half << 3);
            long r   = m0 + m;
            float v  = 0.0f;
            if (r < M && k < kact) {
                if (mode_a == AM_EDGE) {
                    int hd  = kact >> 1;
                    int nd  = (k < hd) ? edst[r] : esrc[r];
                    int c2  = (k < hd) ? k : (k - hd);
                    v = Af[(long)nd * lda + c2];
                } else if (mode_a == AM_F32) {
                    v = Af[r * (long)lda + k];
                } else if (mode_a == AM_BF16) {
                    v = (float)Ab[r * (long)lda + k];
                } else { // AM_BF16_BNACT: fused BatchNorm + LeakySiLU(0.05)
                    float raw = (float)Ab[r * (long)lda + k];
                    v = lsilu(iscale[k] * raw + ishift[k], 0.05f);
                }
            }
            a_lds[i] = (bf16_t)v;
        }
        __syncthreads();

        if (active) {
            v8f acc = {0.f, 0.f, 0.f, 0.f, 0.f, 0.f, 0.f, 0.f};
            for (int kb = 0; kb < KB; ++kb) {
                v16bf a = *(const v16bf*)(a_lds + ((((kb << 4) + arow) * 2 + ahalf) << 4));
                v16bf b = *(const v16bf*)(w_lds + ((((kb << 6) + nl_lane) * 2 + ahalf) << 4));
                acc = __builtin_amdgcn_wmma_f32_16x16x32_bf16(
                          false, a, false, b, (short)0, acc, false, false);
            }
#pragma unroll
            for (int r = 0; r < 8; ++r) {
                long row = m0 + r + (ahalf << 3);
                if (row < M) {
                    float v = acc[r] + bcol;
                    if (epi == EP_BF16_STATS) {
                        Cb[row * (long)ldc + col] = (bf16_t)v;
                        lsum += v; lsq += v * v;
                    } else if (epi == EP_BF16_ACT) {
                        Cb[row * (long)ldc + col] = (bf16_t)lsilu(v, 0.05f);
                    } else {
                        Cf[row * (long)ldc + col] = v;
                        lsum += v; lsq += v * v;
                    }
                }
            }
        }
        __syncthreads();
    }

    if (active && epi != EP_BF16_ACT) {
        atomicAdd(&ssum[col], lsum);
        atomicAdd(&ssq[col],  lsq);
    }
}

// mean/var -> affine scale/shift (training-mode BN, biased variance)
__global__ void k_bn_finalize(const float* __restrict__ ssum, const float* __restrict__ ssq,
                              const float* __restrict__ gamma, const float* __restrict__ beta,
                              float* __restrict__ scale, float* __restrict__ shift,
                              int C, float R) {
    int c = blockIdx.x * blockDim.x + threadIdx.x;
    if (c >= C) return;
    float mean = ssum[c] / R;
    float var  = ssq[c] / R - mean * mean;
    float inv  = rsqrtf(var + 1e-5f);
    float sc   = gamma[c] * inv;
    scale[c] = sc;
    shift[c] = beta[c] - mean * sc;
}

// Per-edge BN2 + LeakySiLU(0.05), atomic scatter-add into node accumulator.
__global__ void k_scatter(const bf16_t* __restrict__ h2,
                          const float* __restrict__ scale, const float* __restrict__ shift,
                          const int* __restrict__ dst, float* __restrict__ acc,
                          long E, int C) {
    long i = (long)blockIdx.x * blockDim.x + threadIdx.x;
    if (i >= E * C) return;
    long e = i / C;
    int  c = (int)(i - e * C);
    float v = lsilu(scale[c] * (float)h2[e * C + c] + shift[c], 0.05f);
    atomicAdd(&acc[(long)dst[e] * 128 + c], v);
}

// nf = LeakySiLU(acc/max(deg,1), 0.1)
__global__ void k_node_finish(const float* __restrict__ acc, const float* __restrict__ deg,
                              float* __restrict__ nf, long NN, int C) {
    long i = (long)blockIdx.x * blockDim.x + threadIdx.x;
    if (i >= NN * C) return;
    long n = i / C;
    int  c = (int)(i - n * C);
    float m = acc[n * 128 + c] / fmaxf(deg[n], 1.0f);
    nf[n * 128 + c] = lsilu(m, 0.1f);
}

// nf += BN(r2)  (residual interlayer)
__global__ void k_inter_res(const float* __restrict__ r2,
                            const float* __restrict__ scale, const float* __restrict__ shift,
                            float* __restrict__ nf, long NN, int C) {
    long i = (long)blockIdx.x * blockDim.x + threadIdx.x;
    if (i >= NN * C) return;
    long n = i / C;
    int  c = (int)(i - n * C);
    nf[n * 128 + c] += scale[c] * r2[n * C + c] + shift[c];
}

// global_add_pool
__global__ void k_pool(const float* __restrict__ nf, const int* __restrict__ bidx,
                       float* __restrict__ g, long NN) {
    long i = (long)blockIdx.x * blockDim.x + threadIdx.x;
    if (i >= NN * 128) return;
    long n = i >> 7;
    int  c = (int)(i & 127);
    atomicAdd(&g[(long)bidx[n] * 128 + c], nf[n * 128 + c]);
}

// FC head: out[b] = (LeakySiLU(g@W1+b1, 0.1) @ W2 + b2) - 100
__global__ __launch_bounds__(128) void k_fc(const float* __restrict__ g,
                                            const float* __restrict__ w1, const float* __restrict__ b1,
                                            const float* __restrict__ w2, const float* __restrict__ b2,
                                            float* __restrict__ out) {
    __shared__ float red[128];
    int b = blockIdx.x, j = threadIdx.x;
    float s = b1[j];
    for (int c = 0; c < 128; ++c) s += g[(long)b * 128 + c] * w1[c * 128 + j];
    s = lsilu(s, 0.1f);
    red[j] = s * w2[j];
    __syncthreads();
    for (int off = 64; off > 0; off >>= 1) {
        if (j < off) red[j] += red[j + off];
        __syncthreads();
    }
    if (j == 0) out[b] = red[0] + b2[0] - 100.0f;
}

// ---------------------------------------------------------------------------
// Host
// ---------------------------------------------------------------------------
static inline int rnd32(int k) { return (k + 31) & ~31; }
static inline long cdivl(long a, long b) { return (a + b - 1) / b; }

static void launch_gemm(hipStream_t s, int mode_a, int epi,
                        const float* Af, const bf16_t* Ab, int lda, int kact,
                        const int* esrc, const int* edst,
                        const float* iscale, const float* ishift,
                        const bf16_t* Wp, const float* bias,
                        bf16_t* Cb, float* Cf, int ldc,
                        float* ssum, float* ssq, long M, int K, int N) {
    dim3 grid((unsigned)cdivl(M, 128), (unsigned)((N + 63) / 64));
    size_t smem = (size_t)K * 160;  // (K*16 + K*64) bf16
    k_gemm<<<grid, 128, smem, s>>>(mode_a, epi, Af, Ab, lda, kact, esrc, edst,
                                   iscale, ishift, Wp, bias, Cb, Cf, ldc,
                                   ssum, ssq, M, K, N);
}

extern "C" void kernel_launch(void* const* d_in, const int* in_sizes, int n_in,
                              void* d_out, int out_size, void* d_ws, size_t ws_size,
                              hipStream_t stream) {
    const int CONV[5][3]  = {{8,256,32},{64,512,128},{256,128,64},{128,256,128},{256,256,128}};
    const int INTER[4]    = {32,128,64,128};

    const float* x      = (const float*)d_in[0];
    const float* matrix = (const float*)d_in[1];
    const int*   bidx   = (const int*)d_in[2];
    const int*   eidx   = (const int*)d_in[3];

    const long NN = in_sizes[0] / 4;
    const int  Bg = in_sizes[1] / 9;
    const long E  = in_sizes[3] / 2;
    const int* esrc = eidx;        // edge_index[0]
    const int* edst = eidx + E;    // edge_index[1]

    int p = 4;
    struct ConvP { const float *w1,*b1,*g1,*be1,*w2,*b2,*g2,*be2; } cv[5];
    for (int c = 0; c < 5; ++c) {
        cv[c].w1 = (const float*)d_in[p++]; cv[c].b1 = (const float*)d_in[p++];
        cv[c].g1 = (const float*)d_in[p++]; cv[c].be1= (const float*)d_in[p++];
        cv[c].w2 = (const float*)d_in[p++]; cv[c].b2 = (const float*)d_in[p++];
        cv[c].g2 = (const float*)d_in[p++]; cv[c].be2= (const float*)d_in[p++];
    }
    struct InterP { const float *w1,*b1,*w2,*b2,*g,*be; } iv[4];
    for (int i = 0; i < 4; ++i) {
        iv[i].w1 = (const float*)d_in[p++]; iv[i].b1 = (const float*)d_in[p++];
        iv[i].w2 = (const float*)d_in[p++]; iv[i].b2 = (const float*)d_in[p++];
        iv[i].g  = (const float*)d_in[p++]; iv[i].be = (const float*)d_in[p++];
    }
    const float* w_fc1 = (const float*)d_in[p++];
    const float* b_fc1 = (const float*)d_in[p++];
    const float* w_fc2 = (const float*)d_in[p++];
    const float* b_fc2 = (const float*)d_in[p++];

    // workspace bump allocator
    char* wsb = (char*)d_ws;
    size_t off = 0;
    auto alloc = [&](size_t bytes) -> void* {
        void* ptr = wsb + off;
        off = (off + bytes + 255) & ~(size_t)255;
        return ptr;
    };

    float* nf   = (float*)alloc((size_t)NN * 128 * 4);
    float* acc  = (float*)alloc((size_t)NN * 128 * 4);
    float* deg  = (float*)alloc((size_t)NN * 4);
    float* gbuf = (float*)alloc((size_t)Bg * 128 * 4);
    float* ssum = (float*)alloc(512 * 4);
    float* ssq  = (float*)alloc(512 * 4);
    float* sc1  = (float*)alloc(512 * 4);
    float* sh1  = (float*)alloc(512 * 4);
    float* sc2  = (float*)alloc(512 * 4);
    float* sh2  = (float*)alloc(512 * 4);
    bf16_t* h1  = (bf16_t*)alloc((size_t)E * 512 * 2);
    bf16_t* h2  = (bf16_t*)alloc((size_t)E * 128 * 2);
    bf16_t* r1  = (bf16_t*)alloc((size_t)NN * 512 * 2);
    float*  r2  = (float*)alloc((size_t)NN * 128 * 4);

    // pack all weights into WMMA fragment layout (bf16)
    bf16_t *cw1[5], *cw2[5], *iw1[4], *iw2[4];
    for (int c = 0; c < 5; ++c) {
        int Kp1 = rnd32(CONV[c][0]);
        cw1[c] = (bf16_t*)alloc((size_t)Kp1 * CONV[c][1] * 2);
        long t1 = (long)Kp1 * CONV[c][1];
        k_pack<<<(unsigned)cdivl(t1, 256), 256, 0, stream>>>(cv[c].w1, cw1[c], CONV[c][0], Kp1, CONV[c][1]);
        int Kp2 = rnd32(CONV[c][1]);
        cw2[c] = (bf16_t*)alloc((size_t)Kp2 * CONV[c][2] * 2);
        long t2 = (long)Kp2 * CONV[c][2];
        k_pack<<<(unsigned)cdivl(t2, 256), 256, 0, stream>>>(cv[c].w2, cw2[c], CONV[c][1], Kp2, CONV[c][2]);
    }
    for (int i = 0; i < 4; ++i) {
        int Kp1 = rnd32(INTER[i]);
        iw1[i] = (bf16_t*)alloc((size_t)Kp1 * 512 * 2);
        k_pack<<<(unsigned)cdivl((long)Kp1 * 512, 256), 256, 0, stream>>>(iv[i].w1, iw1[i], INTER[i], Kp1, 512);
        iw2[i] = (bf16_t*)alloc((size_t)512 * INTER[i] * 2);
        k_pack<<<(unsigned)cdivl((long)512 * INTER[i], 256), 256, 0, stream>>>(iv[i].w2, iw2[i], 512, 512, INTER[i]);
    }

    // node prep + degree
    k_prep<<<(unsigned)cdivl(NN, 256), 256, 0, stream>>>(x, matrix, bidx, nf, NN);
    hipMemsetAsync(deg, 0, (size_t)NN * 4, stream);
    k_degree<<<(unsigned)cdivl(E, 256), 256, 0, stream>>>(edst, deg, E);

    for (int c = 0; c < 5; ++c) {
        const int din = CONV[c][0], dh = CONV[c][1], dout = CONV[c][2];

        // GEMM1: edge gather-concat -> h1 (bf16) + BN1 stats
        hipMemsetAsync(ssum, 0, dh * 4, stream);
        hipMemsetAsync(ssq,  0, dh * 4, stream);
        launch_gemm(stream, AM_EDGE, EP_BF16_STATS, nf, nullptr, 128, din,
                    esrc, edst, nullptr, nullptr, cw1[c], cv[c].b1,
                    h1, nullptr, dh, ssum, ssq, E, rnd32(din), dh);
        k_bn_finalize<<<(dh + 255) / 256, 256, 0, stream>>>(ssum, ssq, cv[c].g1, cv[c].be1,
                                                            sc1, sh1, dh, (float)E);

        // GEMM2: fused BN1+LeakySiLU on load -> h2 (bf16) + BN2 stats
        hipMemsetAsync(ssum, 0, dout * 4, stream);
        hipMemsetAsync(ssq,  0, dout * 4, stream);
        launch_gemm(stream, AM_BF16_BNACT, EP_BF16_STATS, nullptr, h1, dh, dh,
                    nullptr, nullptr, sc1, sh1, cw2[c], cv[c].b2,
                    h2, nullptr, dout, ssum, ssq, E, dh, dout);
        k_bn_finalize<<<(dout + 255) / 256, 256, 0, stream>>>(ssum, ssq, cv[c].g2, cv[c].be2,
                                                              sc2, sh2, dout, (float)E);

        // scatter-mean + LeakySiLU(0.1)
        hipMemsetAsync(acc, 0, (size_t)NN * 128 * 4, stream);
        k_scatter<<<(unsigned)cdivl(E * dout, 256), 256, 0, stream>>>(h2, sc2, sh2, edst, acc, E, dout);
        k_node_finish<<<(unsigned)cdivl(NN * dout, 256), 256, 0, stream>>>(acc, deg, nf, NN, dout);

        if (c < 4) {
            const int d = INTER[c];
            // r1 = LeakySiLU(nf @ W1 + b1, 0.05)   (no BN, so store activated bf16)
            launch_gemm(stream, AM_F32, EP_BF16_ACT, nf, nullptr, 128, d,
                        nullptr, nullptr, nullptr, nullptr, iw1[c], iv[c].b1,
                        r1, nullptr, 512, nullptr, nullptr, NN, rnd32(d), 512);
            // r2 = r1 @ W2 + b2 (fp32) + BN stats over nodes
            hipMemsetAsync(ssum, 0, d * 4, stream);
            hipMemsetAsync(ssq,  0, d * 4, stream);
            launch_gemm(stream, AM_BF16, EP_F32_STATS, nullptr, r1, 512, 512,
                        nullptr, nullptr, nullptr, nullptr, iw2[c], iv[c].b2,
                        nullptr, r2, d, ssum, ssq, NN, 512, d);
            k_bn_finalize<<<1, 256, 0, stream>>>(ssum, ssq, iv[c].g, iv[c].be, sc1, sh1, d, (float)NN);
            k_inter_res<<<(unsigned)cdivl(NN * d, 256), 256, 0, stream>>>(r2, sc1, sh1, nf, NN, d);
        }
    }

    // pool + FC head
    hipMemsetAsync(gbuf, 0, (size_t)Bg * 128 * 4, stream);
    k_pool<<<(unsigned)cdivl(NN * 128, 256), 256, 0, stream>>>(nf, bidx, gbuf, NN);
    k_fc<<<Bg, 128, 0, stream>>>(gbuf, w_fc1, b_fc1, w_fc2, b_fc2, (float*)d_out);

    (void)n_in; (void)out_size; (void)ws_size;
}
